// attention_20151986553082
// MI455X (gfx1250) — matrix operand order repaired and checked by reading, not compiled, not run on hardware
//
#include <hip/hip_runtime.h>
#include <hip/hip_bf16.h>
#include <math.h>

#define BATCH 16
#define IH 56
#define IW 84
#define CH 256
#define NHEAD 8
#define HD 16
#define HWTOK (IH*IW)            // 4704
#define MTOK (BATCH*IH*IW)       // 75264

typedef __attribute__((ext_vector_type(16))) __bf16 v16bf;
typedef __attribute__((ext_vector_type(8)))  float  v8f;

union V16 { uint4 u[2]; v16bf v; };

static __device__ __forceinline__ unsigned short f2bf(float f) {
  unsigned int u = __float_as_uint(f);
  u += 0x7fffu + ((u >> 16) & 1u);          // round-to-nearest-even
  return (unsigned short)(u >> 16);
}
static __device__ __forceinline__ float bf2f(unsigned short h) {
  return __uint_as_float(((unsigned int)h) << 16);
}
static __device__ __forceinline__ float gelu_exact(float x) {
  return 0.5f * x * (1.0f + erff(x * 0.70710678118654752f));
}
static __device__ __forceinline__ int iclamp(int v, int lo, int hi) {
  return v < lo ? lo : (v > hi ? hi : v);
}

// ---------------- weight transpose f32[K][N] -> bf16[N][K] -----------------
__global__ void k_wtrans(const float* __restrict__ w, unsigned short* __restrict__ wt,
                         int K, int N) {
  int i = blockIdx.x * blockDim.x + threadIdx.x;
  if (i >= K * N) return;
  int n = i % N, k = i / N;
  wt[(size_t)n * K + k] = f2bf(w[i]);
}

// ---------------- LayerNorm (one wave per row), f32 in -> bf16 out ---------
__global__ __launch_bounds__(256) void k_layernorm_bf16(
    const float* __restrict__ x, const float* __restrict__ g, const float* __restrict__ b,
    unsigned short* __restrict__ out, int Mrows, int C) {
  int lane = threadIdx.x & 31;
  int wave = threadIdx.x >> 5;
  int row = blockIdx.x * 8 + wave;
  if (row >= Mrows) return;
  const float* xr = x + (size_t)row * C;
  float s = 0.f, s2 = 0.f;
  for (int c = lane; c < C; c += 32) { float v = xr[c]; s += v; s2 += v * v; }
#pragma unroll
  for (int o = 16; o >= 1; o >>= 1) { s += __shfl_xor(s, o, 32); s2 += __shfl_xor(s2, o, 32); }
  float mean = s / C;
  float var = s2 / C - mean * mean;
  float inv = rsqrtf(var + 1e-6f);
  for (int c = lane; c < C; c += 32) {
    float v = (xr[c] - mean) * inv * g[c] + b[c];
    out[(size_t)row * C + c] = f2bf(v);
  }
}

// ---------------- bf16 WMMA GEMM: out = A[M][K] x Wt[N][K]^T + bias --------
// Block = 8 waves => 128(M) x 64(N). The 64xK weight strip (shared by all 8
// waves) is staged once into LDS via gfx1250 async global->LDS copies. Both
// the A fragment (global) and the four B fragments (LDS) are register
// double-buffered one K-step ahead so VMEM/DS latency overlaps the WMMAs.
__global__ __launch_bounds__(256) void k_gemm_wmma_bf16(
    const unsigned short* __restrict__ A, const unsigned short* __restrict__ Wt,
    const float* __restrict__ bias, float* __restrict__ outF,
    unsigned short* __restrict__ outB, int Mrows, int K, int N, int actGelu) {
  extern __shared__ unsigned short sB[];   // [64][K] bf16 strip of Wt
  const int lane = threadIdx.x & 31;
  const int wave = threadIdx.x >> 5;
  const int tileM = blockIdx.x * 8 + wave;
  const int nBase = blockIdx.y * 64;
  const int rl = lane & 15;               // A row / B col / C col
  const int kh = lane >> 4;               // K-half selector

  // ---- async-stage B strip (must run in ALL waves, before any early-out)
  {
    const unsigned short* src = Wt + (size_t)nBase * K;
    const int nchunks = (64 * K) / 8;      // 16-byte chunks
    for (int ch = threadIdx.x; ch < nchunks; ch += 256) {
      unsigned long long g = (unsigned long long)(const void*)(src + (size_t)ch * 8);
      unsigned int l = (unsigned int)(size_t)(const void*)(sB + (size_t)ch * 8);
      asm volatile("global_load_async_to_lds_b128 %0, %1, off"
                   :: "v"(l), "v"(g) : "memory");
    }
    asm volatile("s_wait_asynccnt 0x0" ::: "memory");
    __syncthreads();
  }
  if (tileM * 16 >= Mrows) return;        // wave-uniform: EXEC stays all-ones

  v8f acc[4] = {};
  const unsigned short* aRow = A + (size_t)(tileM * 16 + rl) * K;
  const unsigned short* bRow = sB + (size_t)rl * K + kh * 16;

  // prime the pipeline: fragments for k0 = 0
  V16 af;
  af.u[0] = *(const uint4*)(aRow + kh * 16);       // e0..7  : K = kh*16+0..7
  af.u[1] = *(const uint4*)(aRow + kh * 16 + 8);   // e8..15 : K = kh*16+8..15
  V16 bf[4];
#pragma unroll
  for (int j = 0; j < 4; ++j) {
    const unsigned short* bp = bRow + (size_t)j * 16 * K;
    bf[j].u[0] = *(const uint4*)(bp);
    bf[j].u[1] = *(const uint4*)(bp + 8);
  }

  for (int k0 = 0; k0 < K; k0 += 32) {
    const int kn = (k0 + 32 < K) ? (k0 + 32) : 0;  // branch-free tail prefetch
    V16 afn;
    afn.u[0] = *(const uint4*)(aRow + kn + kh * 16);
    afn.u[1] = *(const uint4*)(aRow + kn + kh * 16 + 8);
    V16 bfn[4];
#pragma unroll
    for (int j = 0; j < 4; ++j) {
      const unsigned short* bp = bRow + (size_t)j * 16 * K + kn;
      bfn[j].u[0] = *(const uint4*)(bp);             // ds_load_b128
      bfn[j].u[1] = *(const uint4*)(bp + 8);         // ds_load_b128
    }
#pragma unroll
    for (int j = 0; j < 4; ++j) {
      acc[j] = __builtin_amdgcn_wmma_f32_16x16x32_bf16(
          false, af.v, false, bf[j].v, (short)0, acc[j], false, false);
    }
    af = afn;
#pragma unroll
    for (int j = 0; j < 4; ++j) bf[j] = bfn[j];
  }

#pragma unroll
  for (int j = 0; j < 4; ++j) {
    int nc = nBase + j * 16 + rl;
    float bv = bias ? bias[nc] : 0.0f;
#pragma unroll
    for (int r = 0; r < 8; ++r) {
      float v = acc[j][r] + bv;
      if (actGelu) v = gelu_exact(v);
      size_t o = (size_t)(tileM * 16 + kh * 8 + r) * N + nc;
      if (outF) outF[o] = v;
      if (outB) outB[o] = f2bf(v);
    }
  }
}

// ---------------- depthwise 7x7 conv, NHWC, bf16 in -> bf16 out ------------
__global__ __launch_bounds__(256) void k_dwconv7_bf16(
    const unsigned short* __restrict__ in, const float* __restrict__ w,
    const float* __restrict__ bias, unsigned short* __restrict__ out, int C) {
  long long i = (long long)blockIdx.x * blockDim.x + threadIdx.x;
  if (i >= (long long)MTOK * C) return;
  int c = (int)(i % C);
  int t = (int)(i / C);
  int b = t / (IH * IW);
  int r = t % (IH * IW);
  int y = r / IW, x = r % IW;
  const float* wc = w + c * 49;
  float acc = bias[c];
  for (int ky = 0; ky < 7; ++ky) {
    int iy = y + ky - 3;
    if (iy < 0 || iy >= IH) continue;
    for (int kx = 0; kx < 7; ++kx) {
      int ix = x + kx - 3;
      if (ix < 0 || ix >= IW) continue;
      acc += wc[ky * 7 + kx] * bf2f(in[(((size_t)(b * IH + iy)) * IW + ix) * C + c]);
    }
  }
  out[i] = f2bf(acc);
}

// ---------------- adaptive avg pool (8x12 blocks) over concat(xn,xen) ------
__global__ void k_pool(const unsigned short* __restrict__ xn,
                       const unsigned short* __restrict__ xen,
                       unsigned short* __restrict__ pooled) {
  int i = blockIdx.x * blockDim.x + threadIdx.x;   // B*49*384
  if (i >= BATCH * 49 * 384) return;
  int c = i % 384;
  int p = (i / 384) % 49;
  int b = i / (49 * 384);
  int py = p / 7, px = p % 7;
  float s = 0.0f;
  for (int dy = 0; dy < 8; ++dy) {
    int y = py * 8 + dy;
    for (int dx = 0; dx < 12; ++dx) {
      int x = px * 12 + dx;
      size_t t = ((size_t)(b * IH + y)) * IW + x;
      s += (c < 256) ? bf2f(xn[t * 256 + c]) : bf2f(xen[t * 128 + (c - 256)]);
    }
  }
  pooled[i] = f2bf(s * (1.0f / 96.0f));
}

// ---------------- pooled attention: online softmax, one wave per query -----
__global__ __launch_bounds__(256) void k_attention(
    const float* __restrict__ mmat, const float* __restrict__ kv,
    float* __restrict__ aosm) {
  int lane = threadIdx.x & 31;
  int wave = threadIdx.x >> 5;
  int rowId = blockIdx.x * 8 + wave;               // B*NH*49 rows
  if (rowId >= BATCH * NHEAD * 49) return;
  int p = rowId % 49;
  int h = (rowId / 49) % NHEAD;
  int b = rowId / (49 * NHEAD);

  float qv[HD];
  const float* mrow = mmat + ((size_t)(b * 49 + p)) * 128 + h * HD;
#pragma unroll
  for (int d = 0; d < HD; ++d) qv[d] = mrow[d] * 0.25f;   // hd^-0.5 = 1/4

  float mx = -3.0e38f, l = 0.f;
  float acc[HD];
#pragma unroll
  for (int d = 0; d < HD; ++d) acc[d] = 0.f;

  const float* kvb = kv + (size_t)b * HWTOK * 256 + h * HD;
  for (int j = lane; j < HWTOK; j += 32) {                // 147 keys per lane
    const float* kp = kvb + (size_t)j * 256;
    __builtin_prefetch(kp + 32 * 256, 0, 1);              // global_prefetch_b8
    float sc = 0.f;
#pragma unroll
    for (int d = 0; d < HD; ++d) sc += qv[d] * kp[d];
    float nm = fmaxf(mx, sc);
    float sca = __expf(mx - nm);
    float e = __expf(sc - nm);
    l = l * sca + e;
    const float* vp = kp + 128;
#pragma unroll
    for (int d = 0; d < HD; ++d) acc[d] = acc[d] * sca + e * vp[d];
    mx = nm;
  }
  // butterfly merge of 32 partial softmax states
#pragma unroll
  for (int o = 16; o >= 1; o >>= 1) {
    float mo = __shfl_xor(mx, o, 32);
    float lo = __shfl_xor(l, o, 32);
    float nm = fmaxf(mx, mo);
    float s0 = __expf(mx - nm), s1 = __expf(mo - nm);
    l = l * s0 + lo * s1;
#pragma unroll
    for (int d = 0; d < HD; ++d) {
      float ad = __shfl_xor(acc[d], o, 32);
      acc[d] = acc[d] * s0 + ad * s1;
    }
    mx = nm;
  }
  if (lane == 0) {
    float inv = 1.0f / l;
#pragma unroll
    for (int d = 0; d < HD; ++d)
      aosm[((size_t)b * 128 + h * HD + d) * 49 + p] = acc[d] * inv;
  }
}

// ---------------- bilinear 7x7 -> 56x84 (half-pixel), to NHWC f32 ----------
__global__ void k_resize(const float* __restrict__ aosm, float* __restrict__ aoup) {
  long long i = (long long)blockIdx.x * blockDim.x + threadIdx.x;  // MTOK*128
  if (i >= (long long)MTOK * 128) return;
  int c = (int)(i % 128);
  int t = (int)(i / 128);
  int b = t / (IH * IW);
  int r = t % (IH * IW);
  int y = r / IW, x = r % IW;
  float fy = (y + 0.5f) * (7.0f / IH) - 0.5f;
  float fx = (x + 0.5f) * (7.0f / IW) - 0.5f;
  float yf = floorf(fy), xf = floorf(fx);
  int y0 = (int)yf, x0 = (int)xf;
  float wy = fy - yf, wx = fx - xf;
  int y0c = iclamp(y0, 0, 6), y1c = iclamp(y0 + 1, 0, 6);
  int x0c = iclamp(x0, 0, 6), x1c = iclamp(x0 + 1, 0, 6);
  const float* chp = aosm + ((size_t)b * 128 + c) * 49;
  float v00 = chp[y0c * 7 + x0c], v01 = chp[y0c * 7 + x1c];
  float v10 = chp[y1c * 7 + x0c], v11 = chp[y1c * 7 + x1c];
  aoup[i] = (1.f - wy) * ((1.f - wx) * v00 + wx * v01)
          +        wy  * ((1.f - wx) * v10 + wx * v11);
}

// ---------------- fuse cat = [q*a | ao | cut*xe2] in bf16 ------------------
__global__ void k_fuse_cat(const float* __restrict__ q, const float* __restrict__ a,
                           const float* __restrict__ aoup, const float* __restrict__ cut,
                           const float* __restrict__ xe2, unsigned short* __restrict__ cat) {
  long long i = (long long)blockIdx.x * blockDim.x + threadIdx.x;  // MTOK*512
  if (i >= (long long)MTOK * 512) return;
  int c = (int)(i % 512);
  size_t t = (size_t)(i / 512);
  float v;
  if (c < 256)      v = q[t * 256 + c] * a[t * 256 + c];
  else if (c < 384) v = aoup[t * 128 + (c - 256)];
  else { int cc = c - 384; v = cut[t * 128 + cc] * xe2[t * 128 + cc]; }
  cat[i] = f2bf(v);
}

extern "C" void kernel_launch(void* const* d_in, const int* in_sizes, int n_in,
                              void* d_out, int out_size, void* d_ws, size_t ws_size,
                              hipStream_t stream) {
  (void)in_sizes; (void)n_in; (void)out_size; (void)ws_size;
  const float* x       = (const float*)d_in[0];
  const float* x_e     = (const float*)d_in[1];
  const float* ln_w    = (const float*)d_in[2];
  const float* ln_b    = (const float*)d_in[3];
  const float* lne_w   = (const float*)d_in[4];
  const float* lne_b   = (const float*)d_in[5];
  const float* q_w     = (const float*)d_in[6];
  const float* q_b     = (const float*)d_in[7];
  const float* qcut_w  = (const float*)d_in[8];
  const float* qcut_b  = (const float*)d_in[9];
  const float* a_w     = (const float*)d_in[10];
  const float* a_b     = (const float*)d_in[11];
  const float* l_w     = (const float*)d_in[12];
  const float* l_b     = (const float*)d_in[13];
  const float* conv_w  = (const float*)d_in[14];
  const float* conv_b  = (const float*)d_in[15];
  const float* econv_w = (const float*)d_in[16];
  const float* econv_b = (const float*)d_in[17];
  const float* efore_w = (const float*)d_in[18];
  const float* efore_b = (const float*)d_in[19];
  const float* eback_w = (const float*)d_in[20];
  const float* eback_b = (const float*)d_in[21];
  const float* kv_w    = (const float*)d_in[22];
  const float* kv_b    = (const float*)d_in[23];
  const float* scl_w   = (const float*)d_in[24];
  const float* scl_b   = (const float*)d_in[25];
  const float* proj_w  = (const float*)d_in[26];
  const float* proj_b  = (const float*)d_in[27];
  const float* proje_w = (const float*)d_in[28];
  const float* proje_b = (const float*)d_in[29];

  char* base = (char*)d_ws;
  size_t off = 0;
  auto alloc = [&](size_t bytes) -> void* {
    void* p = base + off;
    off = (off + bytes + 255) & ~(size_t)255;
    return p;
  };
  // activations
  unsigned short* xn     = (unsigned short*)alloc((size_t)MTOK * 256 * 2);
  unsigned short* xen    = (unsigned short*)alloc((size_t)MTOK * 128 * 2);
  unsigned short* xl     = (unsigned short*)alloc((size_t)MTOK * 256 * 2);
  unsigned short* conva  = (unsigned short*)alloc((size_t)MTOK * 256 * 2);
  unsigned short* efout  = (unsigned short*)alloc((size_t)MTOK * 128 * 2);
  unsigned short* econv  = (unsigned short*)alloc((size_t)MTOK * 128 * 2);
  unsigned short* catb   = (unsigned short*)alloc((size_t)MTOK * 512 * 2);
  float* qf   = (float*)alloc((size_t)MTOK * 256 * 4);
  float* af   = (float*)alloc((size_t)MTOK * 256 * 4);
  float* cutf = (float*)alloc((size_t)MTOK * 128 * 4);
  float* kvf  = (float*)alloc((size_t)MTOK * 256 * 4);
  float* xe2f = (float*)alloc((size_t)MTOK * 128 * 4);
  float* aoup = (float*)alloc((size_t)MTOK * 128 * 4);
  unsigned short* pooled = (unsigned short*)alloc((size_t)BATCH * 49 * 384 * 2);
  float* mmat = (float*)alloc((size_t)BATCH * 49 * 128 * 4);
  float* aosm = (float*)alloc((size_t)BATCH * 128 * 49 * 4);
  // transposed bf16 weights
  unsigned short* wq    = (unsigned short*)alloc((size_t)256 * 256 * 2);
  unsigned short* wqcut = (unsigned short*)alloc((size_t)256 * 128 * 2);
  unsigned short* wa    = (unsigned short*)alloc((size_t)256 * 256 * 2);
  unsigned short* wl    = (unsigned short*)alloc((size_t)256 * 256 * 2);
  unsigned short* wkv   = (unsigned short*)alloc((size_t)256 * 256 * 2);
  unsigned short* wef   = (unsigned short*)alloc((size_t)128 * 128 * 2);
  unsigned short* web   = (unsigned short*)alloc((size_t)128 * 128 * 2);
  unsigned short* wscl  = (unsigned short*)alloc((size_t)384 * 128 * 2);
  unsigned short* wproj = (unsigned short*)alloc((size_t)512 * 256 * 2);
  unsigned short* wproje= (unsigned short*)alloc((size_t)512 * 128 * 2);

  auto cdiv = [](long long a, long long b) { return (int)((a + b - 1) / b); };

  // 1) weights -> bf16 transposed [N][K]
  k_wtrans<<<cdiv(256 * 256, 256), 256, 0, stream>>>(q_w, wq, 256, 256);
  k_wtrans<<<cdiv(256 * 128, 256), 256, 0, stream>>>(qcut_w, wqcut, 256, 128);
  k_wtrans<<<cdiv(256 * 256, 256), 256, 0, stream>>>(a_w, wa, 256, 256);
  k_wtrans<<<cdiv(256 * 256, 256), 256, 0, stream>>>(l_w, wl, 256, 256);
  k_wtrans<<<cdiv(256 * 256, 256), 256, 0, stream>>>(kv_w, wkv, 256, 256);
  k_wtrans<<<cdiv(128 * 128, 256), 256, 0, stream>>>(efore_w, wef, 128, 128);
  k_wtrans<<<cdiv(128 * 128, 256), 256, 0, stream>>>(eback_w, web, 128, 128);
  k_wtrans<<<cdiv(384 * 128, 256), 256, 0, stream>>>(scl_w, wscl, 384, 128);
  k_wtrans<<<cdiv(512 * 256, 256), 256, 0, stream>>>(proj_w, wproj, 512, 256);
  k_wtrans<<<cdiv(512 * 128, 256), 256, 0, stream>>>(proje_w, wproje, 512, 128);

  // 2) LayerNorms
  k_layernorm_bf16<<<MTOK / 8, 256, 0, stream>>>(x, ln_w, ln_b, xn, MTOK, 256);
  k_layernorm_bf16<<<MTOK / 8, 256, 0, stream>>>(x_e, lne_w, lne_b, xen, MTOK, 128);

  // 3) big GEMMs (WMMA, LDS-staged B)  -- dynamic LDS = 64*K*2 bytes
  dim3 g256(MTOK / 16 / 8, 4), g128(MTOK / 16 / 8, 2);
  k_gemm_wmma_bf16<<<g256, 256, 64 * 256 * 2, stream>>>(xn, wq, q_b, qf, nullptr, MTOK, 256, 256, 0);
  k_gemm_wmma_bf16<<<g128, 256, 64 * 256 * 2, stream>>>(xn, wqcut, qcut_b, cutf, nullptr, MTOK, 256, 128, 0);
  k_gemm_wmma_bf16<<<g256, 256, 64 * 256 * 2, stream>>>(xn, wl, l_b, nullptr, xl, MTOK, 256, 256, 1); // GELU

  // 4) depthwise conv (a-branch) + a / kv GEMMs
  k_dwconv7_bf16<<<cdiv((long long)MTOK * 256, 256), 256, 0, stream>>>(xl, conv_w, conv_b, conva, 256);
  k_gemm_wmma_bf16<<<g256, 256, 64 * 256 * 2, stream>>>(conva, wa, a_b, af, nullptr, MTOK, 256, 256, 0);
  k_gemm_wmma_bf16<<<g256, 256, 64 * 256 * 2, stream>>>(xl, wkv, kv_b, kvf, nullptr, MTOK, 256, 256, 0);

  // 5) pooled attention branch
  k_pool<<<cdiv(BATCH * 49 * 384, 256), 256, 0, stream>>>(xn, xen, pooled);
  dim3 gscl(cdiv(784 / 16, 8), 2);
  k_gemm_wmma_bf16<<<gscl, 256, 64 * 384 * 2, stream>>>(pooled, wscl, scl_b, mmat, nullptr, 784, 384, 128, 0);
  k_attention<<<cdiv(BATCH * NHEAD * 49, 8), 256, 0, stream>>>(mmat, kvf, aosm);
  k_resize<<<cdiv((long long)MTOK * 128, 256), 256, 0, stream>>>(aosm, aoup);

  // 6) depth branch
  k_gemm_wmma_bf16<<<g128, 256, 64 * 128 * 2, stream>>>(xen, wef, efore_b, nullptr, efout, MTOK, 128, 128, 0);
  k_dwconv7_bf16<<<cdiv((long long)MTOK * 128, 256), 256, 0, stream>>>(efout, econv_w, econv_b, econv, 128);
  k_gemm_wmma_bf16<<<g128, 256, 64 * 128 * 2, stream>>>(econv, web, eback_b, xe2f, nullptr, MTOK, 128, 128, 0);

  // 7) fuse + output projections
  k_fuse_cat<<<cdiv((long long)MTOK * 512, 256), 256, 0, stream>>>(qf, af, aoup, cutf, xe2f, catb);
  float* out_x = (float*)d_out;
  float* out_xe = out_x + (size_t)MTOK * 256;
  k_gemm_wmma_bf16<<<g256, 256, 64 * 512 * 2, stream>>>(catb, wproj, proj_b, out_x, nullptr, MTOK, 512, 256, 0);
  k_gemm_wmma_bf16<<<g128, 256, 64 * 512 * 2, stream>>>(catb, wproje, proje_b, out_xe, nullptr, MTOK, 512, 128, 0);
}
